// GAT_25177098289354
// MI455X (gfx1250) — matrix operand order repaired and checked by reading, not compiled, not run on hardware
//
#include <hip/hip_runtime.h>

typedef __attribute__((ext_vector_type(16))) __bf16 v16bf;
typedef __attribute__((ext_vector_type(8)))  __bf16 v8bf;
typedef __attribute__((ext_vector_type(8)))  float  v8f;

#define NN 50000
#define NE 800000
#define IND 256
#define HID 32
#define NH 4
#define OUTD 64

// ---------------- helpers ----------------

__device__ __forceinline__ unsigned fenc(float f) {
    unsigned u = __float_as_uint(f);
    return (u & 0x80000000u) ? ~u : (u | 0x80000000u);
}
__device__ __forceinline__ float fdec(unsigned u) {
    unsigned v = (u & 0x80000000u) ? (u & 0x7FFFFFFFu) : ~u;
    return __uint_as_float(v);
}

// ---------------- conversion / packing ----------------

__global__ void cvt_bf16_kernel(const float* __restrict__ x, __bf16* __restrict__ y, int n) {
    int i = blockIdx.x * blockDim.x + threadIdx.x;
    if (i < n) y[i] = (__bf16)x[i];
}

// W[h][k][d] (f32) -> Wt[n][k] (bf16), n = h*D + d
__global__ void pack_w_kernel(const float* __restrict__ W, __bf16* __restrict__ Wt,
                              int H, int K, int D) {
    int i = blockIdx.x * blockDim.x + threadIdx.x;
    if (i >= H * D * K) return;
    int n = i / K, k = i % K;
    int h = n / D, d = n % D;
    Wt[i] = (__bf16)W[((size_t)h * K + k) * D + d];
}

// ---------------- WMMA projection GEMM ----------------
// Z[M x N] = X[M x K] (bf16) * Wt^T  where Wt is [N x K] bf16 (row = output channel).
// One wave computes one 16x16 D tile; 4 waves per block cover 64 columns.
template<int K, int N>
__global__ __launch_bounds__(128) void gemm_wmma_kernel(const __bf16* __restrict__ X,
                                                        const __bf16* __restrict__ Wt,
                                                        float* __restrict__ Z) {
    const int lane = threadIdx.x & 31;
    const int wv   = threadIdx.x >> 5;
    const int nt   = blockIdx.x * 4 + wv;   // N/16 tiles, exact
    const int mt   = blockIdx.y;            // M/16 tiles, exact (50000 = 3125*16)
    const int lo   = lane & 15;
    const int hi   = lane >> 4;

    // A fragment: lane holds row m = lo; elems 0..7 -> K = 8*hi + j, elems 8..15 -> K = 16 + 8*hi + j
    const __bf16* arow = X  + (size_t)(mt * 16 + lo) * K + 8 * hi;
    // B fragment: lane holds col n = lo; elems 0..7 -> K = 8*hi + j, elems 8..15 -> K = 16 + 8*hi + j  (Wt row-major over K)
    const __bf16* brow = Wt + (size_t)(nt * 16 + lo) * K + 8 * hi;

    v8f acc = {};
#pragma unroll
    for (int k0 = 0; k0 < K; k0 += 32) {
        v8bf a0 = *(const v8bf*)(arow + k0);
        v8bf a1 = *(const v8bf*)(arow + k0 + 16);
        v8bf b0 = *(const v8bf*)(brow + k0);
        v8bf b1 = *(const v8bf*)(brow + k0 + 16);
        v16bf a, b;
#pragma unroll
        for (int i = 0; i < 8; ++i) {
            a[i] = a0[i]; a[8 + i] = a1[i];
            b[i] = b0[i]; b[8 + i] = b1[i];
        }
        acc = __builtin_amdgcn_wmma_f32_16x16x32_bf16(
            /*neg_a=*/false, a, /*neg_b=*/false, b,
            /*c_mod=*/(short)0, acc, /*reuse_a=*/false, /*reuse_b=*/false);
    }
    // D layout: VGPR r -> M = r + 8*hi, N = lo
    float* zcol = Z + (size_t)(mt * 16 + 8 * hi) * N + nt * 16 + lo;
#pragma unroll
    for (int r = 0; r < 8; ++r) zcol[(size_t)r * N] = acc[r];
}

// ---------------- attention scores ----------------
// es[n,h] = sum_o z[n,h,o]*a[h,o] ; ed[n,h] = sum_o z[n,h,o]*a[h,D+o]
template<int H, int D>
__global__ void scores_kernel(const float* __restrict__ z, const float* __restrict__ a,
                              float* __restrict__ es, float* __restrict__ ed) {
    int i = blockIdx.x * blockDim.x + threadIdx.x;
    if (i >= NN * H) return;
    int n = i / H, h = i % H;
    const float* zr = z + (size_t)n * (H * D) + h * D;
    const float* as = a + h * 2 * D;
    const float* ad = as + D;
    float s = 0.f, t = 0.f;
#pragma unroll 8
    for (int o = 0; o < D; ++o) { s += zr[o] * as[o]; t += zr[o] * ad[o]; }
    es[i] = s; ed[i] = t;
}

// ---------------- edge softmax pass 1: leaky_relu + segment max ----------------
template<int H>
__global__ void edge_max_kernel(const int* __restrict__ src, const int* __restrict__ dst,
                                const float* __restrict__ es, const float* __restrict__ ed,
                                float* __restrict__ ee, unsigned* __restrict__ emax) {
    int i = blockIdx.x * blockDim.x + threadIdx.x;
    if (i >= NE * H) return;
    int e = i / H, h = i % H;
    float v = es[src[e] * H + h] + ed[dst[e] * H + h];
    v = v > 0.f ? v : 0.01f * v;   // jax.nn.leaky_relu default slope 0.01
    ee[i] = v;
    atomicMax(&emax[dst[e] * H + h], fenc(v));
}

// ---------------- edge softmax pass 2: exp + segment sum ----------------
template<int H>
__global__ void edge_exp_kernel(const int* __restrict__ dst, float* __restrict__ ee,
                                const unsigned* __restrict__ emax, float* __restrict__ denom) {
    int i = blockIdx.x * blockDim.x + threadIdx.x;
    if (i >= NE * H) return;
    int e = i / H, h = i % H;
    float ex = __expf(ee[i] - fdec(emax[dst[e] * H + h]));
    ee[i] = ex;
    atomicAdd(&denom[dst[e] * H + h], ex);
}

// ---------------- aggregation: out[dst] += alpha * z[src] ----------------
// one wave per edge; lanes striped across H*D features (coalesced, L2-resident atomics)
template<int H, int D>
__global__ __launch_bounds__(256) void agg_kernel(const int* __restrict__ src,
                                                  const int* __restrict__ dst,
                                                  const float* __restrict__ z,
                                                  const float* __restrict__ ee,
                                                  const float* __restrict__ denom,
                                                  float* __restrict__ out) {
    const int NF = H * D;
    int gid  = blockIdx.x * blockDim.x + threadIdx.x;
    int e    = gid >> 5;
    int lane = gid & 31;
    if (e >= NE) return;
    int s = src[e], t = dst[e];
#pragma unroll
    for (int j = 0; j < NF / 32; ++j) {
        int idx = j * 32 + lane;
        int h = idx / D;
        float alpha = ee[(size_t)e * H + h] / denom[t * H + h];
        atomicAdd(&out[(size_t)t * NF + idx], alpha * z[(size_t)s * NF + idx]);
    }
}

// ---------------- ELU + bf16 cast between layers ----------------
__global__ void elu_bf16_kernel(const float* __restrict__ x, __bf16* __restrict__ y, int n) {
    int i = blockIdx.x * blockDim.x + threadIdx.x;
    if (i < n) {
        float v = x[i];
        float r = v > 0.f ? v : (__expf(v) - 1.f);
        y[i] = (__bf16)r;
    }
}

// ---------------- host orchestration ----------------

extern "C" void kernel_launch(void* const* d_in, const int* in_sizes, int n_in,
                              void* d_out, int out_size, void* d_ws, size_t ws_size,
                              hipStream_t stream) {
    const float* h   = (const float*)d_in[0];
    const int*   src = (const int*)d_in[1];
    const int*   dst = (const int*)d_in[2];
    const float* W1  = (const float*)d_in[3];
    const float* a1  = (const float*)d_in[4];
    const float* W2  = (const float*)d_in[5];
    const float* a2  = (const float*)d_in[6];
    float* out = (float*)d_out;

    char* base = (char*)d_ws;
    size_t off = 0;
    auto alloc = [&](size_t bytes) -> char* {
        char* r = base + off;
        off += (bytes + 255) & ~(size_t)255;
        return r;
    };

    const int N1 = NH * HID;   // 128
    const int N2 = OUTD;       // 64

    __bf16*   hbf    = (__bf16*)  alloc((size_t)NN * IND * 2);
    __bf16*   Wt1    = (__bf16*)  alloc((size_t)N1 * IND * 2);
    __bf16*   Wt2    = (__bf16*)  alloc((size_t)N2 * N1 * 2);
    float*    z1     = (float*)   alloc((size_t)NN * N1 * 4);
    float*    es1    = (float*)   alloc((size_t)NN * NH * 4);
    float*    ed1    = (float*)   alloc((size_t)NN * NH * 4);
    unsigned* emax1  = (unsigned*)alloc((size_t)NN * NH * 4);
    float*    denom1 = (float*)   alloc((size_t)NN * NH * 4);
    float*    ee1    = (float*)   alloc((size_t)NE * NH * 4);
    float*    out1   = (float*)   alloc((size_t)NN * N1 * 4);
    __bf16*   h1bf   = (__bf16*)  alloc((size_t)NN * N1 * 2);
    float*    z2     = (float*)   alloc((size_t)NN * N2 * 4);
    float*    es2    = (float*)   alloc((size_t)NN * 4);
    float*    ed2    = (float*)   alloc((size_t)NN * 4);
    unsigned* emax2  = (unsigned*)alloc((size_t)NN * 4);
    float*    denom2 = (float*)   alloc((size_t)NN * 4);
    float*    ee2    = (float*)   alloc((size_t)NE * 4);

    // zero-init accumulators (graph-capture-safe stream memsets)
    hipMemsetAsync(emax1,  0, (size_t)NN * NH * 4, stream);
    hipMemsetAsync(denom1, 0, (size_t)NN * NH * 4, stream);
    hipMemsetAsync(out1,   0, (size_t)NN * N1 * 4, stream);
    hipMemsetAsync(emax2,  0, (size_t)NN * 4, stream);
    hipMemsetAsync(denom2, 0, (size_t)NN * 4, stream);
    hipMemsetAsync(d_out,  0, (size_t)out_size * 4, stream);

    // input conversions / weight packing
    cvt_bf16_kernel<<<(NN * IND + 255) / 256, 256, 0, stream>>>(h, hbf, NN * IND);
    pack_w_kernel<<<(N1 * IND + 255) / 256, 256, 0, stream>>>(W1, Wt1, NH, IND, HID);
    pack_w_kernel<<<(N2 * N1 + 255) / 256, 256, 0, stream>>>(W2, Wt2, 1, N1, OUTD);

    // ---- layer 1 ----
    gemm_wmma_kernel<IND, 128><<<dim3(128 / 64, NN / 16), 128, 0, stream>>>(hbf, Wt1, z1);
    scores_kernel<NH, HID><<<(NN * NH + 255) / 256, 256, 0, stream>>>(z1, a1, es1, ed1);
    edge_max_kernel<NH><<<((size_t)NE * NH + 255) / 256, 256, 0, stream>>>(src, dst, es1, ed1, ee1, emax1);
    edge_exp_kernel<NH><<<((size_t)NE * NH + 255) / 256, 256, 0, stream>>>(dst, ee1, emax1, denom1);
    agg_kernel<NH, HID><<<((size_t)NE * 32 + 255) / 256, 256, 0, stream>>>(src, dst, z1, ee1, denom1, out1);
    elu_bf16_kernel<<<(NN * N1 + 255) / 256, 256, 0, stream>>>(out1, h1bf, NN * N1);

    // ---- layer 2 ----
    gemm_wmma_kernel<128, 64><<<dim3(64 / 64, NN / 16), 128, 0, stream>>>(h1bf, Wt2, z2);
    scores_kernel<1, OUTD><<<(NN + 255) / 256, 256, 0, stream>>>(z2, a2, es2, ed2);
    edge_max_kernel<1><<<(NE + 255) / 256, 256, 0, stream>>>(src, dst, es2, ed2, ee2, emax2);
    edge_exp_kernel<1><<<(NE + 255) / 256, 256, 0, stream>>>(dst, ee2, emax2, denom2);
    agg_kernel<1, OUTD><<<((size_t)NE * 32 + 255) / 256, 256, 0, stream>>>(src, dst, z2, ee2, denom2, out);
}
